// CagnetSAGE_8452495639005
// MI455X (gfx1250) — matrix operand order repaired and checked by reading, not compiled, not run on hardware
//
#include <hip/hip_runtime.h>
#include <hip/hip_bf16.h>

#define D      128
#define TILE_M 16
#define LDA    (D + 8)   // padded LDS row stride in halfs (272 B) -> conflict-free b128 reads

typedef __attribute__((ext_vector_type(16))) __bf16 v16bf;
typedef __attribute__((ext_vector_type(8)))  float  v8f;

__device__ __forceinline__ unsigned short f2bf(float f) {
  union { float f; unsigned u; } x; x.f = f;
  unsigned u = x.u + 0x7FFFu + ((x.u >> 16) & 1u);   // round-to-nearest-even
  return (unsigned short)(u >> 16);
}

__device__ __forceinline__ void atomic_add_f32(float* p, float v) {
  __hip_atomic_fetch_add(p, v, __ATOMIC_RELAXED, __HIP_MEMORY_SCOPE_AGENT);
}

// ---------------------------------------------------------------------------
// Convert 4 weight matrices (128x128 f32, row-major [n][k]) to bf16, and
// build combined biases: bias[0:128]=bn1+bs1, bias[128:256]=bn2+bs2.
// ---------------------------------------------------------------------------
__global__ void sage_prep_weights(const float* __restrict__ Wn1, const float* __restrict__ Ws1,
                                  const float* __restrict__ Wn2, const float* __restrict__ Ws2,
                                  const float* __restrict__ bn1, const float* __restrict__ bs1,
                                  const float* __restrict__ bn2, const float* __restrict__ bs2,
                                  unsigned short* __restrict__ Wbf, float* __restrict__ bias) {
  int idx = blockIdx.x * blockDim.x + threadIdx.x;   // 0 .. 4*128*128-1
  int m   = idx >> 14;                               // which matrix
  int off = idx & 16383;
  const float* s = (m == 0) ? Wn1 : (m == 1) ? Ws1 : (m == 2) ? Wn2 : Ws2;
  Wbf[idx] = f2bf(s[off]);
  if (idx < D) {
    bias[idx]     = bn1[idx] + bs1[idx];
    bias[D + idx] = bn2[idx] + bs2[idx];
  }
}

// ---------------------------------------------------------------------------
// Degree / inverse-degree
// ---------------------------------------------------------------------------
__global__ void sage_degree(const int* __restrict__ dst, float* __restrict__ deg, int E) {
  int e = blockIdx.x * blockDim.x + threadIdx.x;
  if (e < E) atomic_add_f32(&deg[dst[e]], 1.0f);
}

__global__ void sage_invdeg(float* __restrict__ deg, int n) {
  int i = blockIdx.x * blockDim.x + threadIdx.x;
  if (i < n) deg[i] = 1.0f / fmaxf(deg[i], 1.0f);
}

// ---------------------------------------------------------------------------
// Mean-aggregation scatter: one wave per edge, lane l moves float4 l of the
// 128-float feature row (512 B fully coalesced), atomic f32 adds to Agg[dst].
// ---------------------------------------------------------------------------
__global__ __launch_bounds__(256) void sage_scatter(const float* __restrict__ H,
                                                    const int* __restrict__ src,
                                                    const int* __restrict__ dst,
                                                    float* __restrict__ Agg, int E) {
  int gid  = blockIdx.x * blockDim.x + threadIdx.x;
  int e    = gid >> 5;
  int lane = gid & 31;
  if (e >= E) return;
  int s = src[e], d = dst[e];
  const float4 v = *(const float4*)(H + (size_t)s * D + lane * 4);
  float* o = Agg + (size_t)d * D + lane * 4;
  atomic_add_f32(o + 0, v.x);
  atomic_add_f32(o + 1, v.y);
  atomic_add_f32(o + 2, v.z);
  atomic_add_f32(o + 3, v.w);
}

// ---------------------------------------------------------------------------
// Fused SAGE layer: Out = act( (Agg*inv_deg) @ Wn^T + Hself @ Ws^T + bias )
// Block = 256 threads = 8 waves; block handles one 16-row M tile, wave w
// handles column tile w (16 cols). K=128 as 4 steps of v_wmma_f32_16x16x32_bf16.
// ---------------------------------------------------------------------------
template <bool RELU>
__global__ __launch_bounds__(256) void sage_layer(const float* __restrict__ Hself,
                                                  const float* __restrict__ Agg,
                                                  const float* __restrict__ invdeg,
                                                  const unsigned short* __restrict__ Wn,
                                                  const unsigned short* __restrict__ Ws,
                                                  const float* __restrict__ bias,
                                                  float* __restrict__ Out, int n_nodes) {
  __shared__ unsigned short sAgg[TILE_M][LDA];
  __shared__ unsigned short sX[TILE_M][LDA];

  const int tid   = threadIdx.x;
  const int mbase = blockIdx.x * TILE_M;

  // ---- stage A tiles (scaled agg + self) into LDS as bf16: 8 elems/thread ----
  {
    int idx = tid * 8;                 // 0..2040
    int r   = idx >> 7;                // row within tile (16 threads per row)
    int c   = idx & 127;
    int gr  = mbase + r; if (gr >= n_nodes) gr = n_nodes - 1;
    size_t g = (size_t)gr * D + c;
    float  sc = invdeg[gr];
    float4 a0 = *(const float4*)(Agg + g);
    float4 a1 = *(const float4*)(Agg + g + 4);
    float4 x0 = *(const float4*)(Hself + g);
    float4 x1 = *(const float4*)(Hself + g + 4);
    unsigned short* pa = &sAgg[r][c];
    pa[0] = f2bf(a0.x * sc); pa[1] = f2bf(a0.y * sc);
    pa[2] = f2bf(a0.z * sc); pa[3] = f2bf(a0.w * sc);
    pa[4] = f2bf(a1.x * sc); pa[5] = f2bf(a1.y * sc);
    pa[6] = f2bf(a1.z * sc); pa[7] = f2bf(a1.w * sc);
    unsigned short* px = &sX[r][c];
    px[0] = f2bf(x0.x); px[1] = f2bf(x0.y); px[2] = f2bf(x0.z); px[3] = f2bf(x0.w);
    px[4] = f2bf(x1.x); px[5] = f2bf(x1.y); px[6] = f2bf(x1.z); px[7] = f2bf(x1.w);
  }
  __syncthreads();

  const int wave = tid >> 5;     // column tile 0..7
  const int lane = tid & 31;
  const int nIdx = lane & 15;    // A row (M) / B column (N) / C column
  const int half = lane >> 4;

  union Frag { v16bf v; uint4 q[2]; };
  v8f acc = {};

  const unsigned short* wn = Wn + (size_t)(wave * 16 + nIdx) * D;
  const unsigned short* ws = Ws + (size_t)(wave * 16 + nIdx) * D;

#pragma unroll
  for (int k = 0; k < 4; ++k) {
    // A fragment: lane L holds row L%16; halfs j=0..7 -> K = 8h..8h+7,
    // j=8..15 -> K = 16+8h..23+8h  (ISA 16-bit A 16x32 layout)
    const int ka = k * 32 + half * 8;
    Frag aA, aX;
    aA.q[0] = *(const uint4*)&sAgg[nIdx][ka];
    aA.q[1] = *(const uint4*)&sAgg[nIdx][ka + 16];
    aX.q[0] = *(const uint4*)&sX[nIdx][ka];
    aX.q[1] = *(const uint4*)&sX[nIdx][ka + 16];
    // B fragment: lane L holds col L%16; halfs j=0..7 -> K = 8h..8h+7, j=8..15 -> K = 16+8h..23+8h
    const int kb = k * 32 + half * 8;
    Frag bN, bS;
    bN.q[0] = *(const uint4*)(wn + kb);
    bN.q[1] = *(const uint4*)(wn + kb + 16);
    bS.q[0] = *(const uint4*)(ws + kb);
    bS.q[1] = *(const uint4*)(ws + kb + 16);

    acc = __builtin_amdgcn_wmma_f32_16x16x32_bf16(false, aA.v, false, bN.v,
                                                  (short)0, acc, false, false);
    acc = __builtin_amdgcn_wmma_f32_16x16x32_bf16(false, aX.v, false, bS.v,
                                                  (short)0, acc, false, false);
  }

  // ---- epilogue: bias + activation; C lane L: col L%16, VGPR r -> row r+8h ----
  const float b = bias[wave * 16 + nIdx];
#pragma unroll
  for (int r = 0; r < 8; ++r) {
    int row = mbase + r + half * 8;
    if (row < n_nodes) {
      float v = acc[r] + b;
      if (RELU) v = fmaxf(v, 0.0f);
      Out[(size_t)row * D + wave * 16 + nIdx] = v;
    }
  }
}

// ---------------------------------------------------------------------------
extern "C" void kernel_launch(void* const* d_in, const int* in_sizes, int n_in,
                              void* d_out, int out_size, void* d_ws, size_t ws_size,
                              hipStream_t stream) {
  const float* x   = (const float*)d_in[0];
  const int*   ei  = (const int*)d_in[1];
  const float* Wn1 = (const float*)d_in[2];
  const float* bn1 = (const float*)d_in[3];
  const float* Ws1 = (const float*)d_in[4];
  const float* bs1 = (const float*)d_in[5];
  const float* Wn2 = (const float*)d_in[6];
  const float* bn2 = (const float*)d_in[7];
  const float* Ws2 = (const float*)d_in[8];
  const float* bs2 = (const float*)d_in[9];

  const int n = in_sizes[0] / D;     // 100000
  const int E = in_sizes[1] / 2;     // 1.6M
  const int* src = ei;
  const int* dst = ei + E;

  // workspace layout
  char* ws = (char*)d_ws;
  float* deg = (float*)ws;            ws += (((size_t)n * 4) + 255) & ~(size_t)255;
  float* agg = (float*)ws;            ws += (size_t)n * D * 4;
  float* h1  = (float*)ws;            ws += (size_t)n * D * 4;
  unsigned short* Wbf = (unsigned short*)ws;  ws += (size_t)4 * D * D * 2;
  float* bias = (float*)ws;           ws += (size_t)2 * D * 4;

  hipMemsetAsync(deg, 0, (size_t)n * 4, stream);
  hipMemsetAsync(agg, 0, (size_t)n * D * 4, stream);

  sage_prep_weights<<<(4 * D * D) / 256, 256, 0, stream>>>(Wn1, Ws1, Wn2, Ws2,
                                                           bn1, bs1, bn2, bs2, Wbf, bias);
  sage_degree<<<(E + 255) / 256, 256, 0, stream>>>(dst, deg, E);
  sage_invdeg<<<(n + 255) / 256, 256, 0, stream>>>(deg, n);

  const int scatter_blocks = (E * 32 + 255) / 256;   // one wave per edge
  const int gemm_blocks    = (n + TILE_M - 1) / TILE_M;

  // layer 1: agg(x) -> h1 = relu(...)
  sage_scatter<<<scatter_blocks, 256, 0, stream>>>(x, src, dst, agg, E);
  sage_layer<true><<<gemm_blocks, 256, 0, stream>>>(x, agg, deg, Wbf, Wbf + D * D,
                                                    bias, h1, n);

  // layer 2: agg(h1) -> out (no relu)
  hipMemsetAsync(agg, 0, (size_t)n * D * 4, stream);
  sage_scatter<<<scatter_blocks, 256, 0, stream>>>(h1, src, dst, agg, E);
  sage_layer<false><<<gemm_blocks, 256, 0, stream>>>(h1, agg, deg, Wbf + 2 * D * D,
                                                     Wbf + 3 * D * D, bias + D,
                                                     (float*)d_out, n);
}
